// GHGNN_model_20203526160535
// MI455X (gfx1250) — compile-verified, matrix operand order checked by reading, not run on hardware
//
#include <hip/hip_runtime.h>
#include <hip/hip_bf16.h>
#include <math.h>

// ---------------------------------------------------------------------------
// GHGNN forward for MI455X (gfx1250): all dense layers via v_wmma_f32_16x16x32_bf16
// - Fragments: two aligned ds_load_b128 per operand (chunk-XOR bank swizzle)
// - Staging: branch-free main loop (M,N are multiples of 64 for every call),
//   float4 global loads + ds_store_b64 when K%4==0, uniform scalar K-tail.
// ---------------------------------------------------------------------------

#define HID  128
#define VIN  64
#define EIN  16
#define UIN  3
#define BG   512         // graphs per batch
#define NPG  64          // nodes per graph
#define EPG  128         // edges per graph
#define NN   (BG * NPG)  // 32768 nodes
#define NE   (BG * EPG)  // 65536 edges
#define H2   256
#define EHD  32

typedef __attribute__((ext_vector_type(16))) __bf16 bhalf16;
typedef __attribute__((ext_vector_type(8)))  float  float8;

struct alignas(16) U4 { unsigned int x, y, z, w; };
struct alignas(16) F4 { float x, y, z, w; };
union FragU { U4 q[2]; bhalf16 v; };

static __device__ inline unsigned int packbf(float a, float b) {
    union { __bf16 h[2]; unsigned int u; } q;
    q.h[0] = (__bf16)a;
    q.h[1] = (__bf16)b;
    return q.u;
}

// ---------------------------------------------------------------------------
// Generic GEMM:  C[M,N] (+)= act( A[M,K] @ W + bias )
//   W row-major [K,N] (TRANSW=0) or [N,K] (TRANSW=1, i.e. A @ W^T)
// REQUIRES: M % 64 == 0 and N % 64 == 0 (true for every call in this model).
// 128 threads = 4 wave32; block tile 64x64; K stepped by 32 through LDS bf16.
// ---------------------------------------------------------------------------
template <int TRANSW, int RELU, int ACCUM>
__global__ __launch_bounds__(128)
void gemm_bf16_wmma(const float* __restrict__ A, const float* __restrict__ W,
                    const float* __restrict__ bias, float* __restrict__ C,
                    int M, int N, int K) {
    __shared__ alignas(16) unsigned int As[64][16];  // [row][kword]  (swizzled)
    __shared__ alignas(16) unsigned int Bs[64][16];  // [col][kword]  (swizzled)

    const int tid  = threadIdx.x;
    const int wv   = tid >> 5;      // wave 0..3
    const int lane = tid & 31;
    const int hf   = lane >> 4;     // lane half
    const int lm   = lane & 15;
    const int m0   = blockIdx.x * 64;
    const int n0   = blockIdx.y * 64;

    float8 acc[4];
#pragma unroll
    for (int t = 0; t < 4; ++t)
#pragma unroll
        for (int j = 0; j < 8; ++j) acc[t][j] = 0.0f;

    // fragment-load + WMMA body (reused by main loop and K-tail)
    const int arow = wv * 16 + lm;
    const int asw  = (arow >> 2) & 3;
    auto mma_step = [&]() {
        __syncthreads();
        FragU fa;
        fa.q[0] = *(const U4*)&As[arow][(hf ^ asw) << 2];
        fa.q[1] = *(const U4*)&As[arow][((2 + hf) ^ asw) << 2];
#pragma unroll
        for (int t = 0; t < 4; ++t) {
            const int bcol = t * 16 + lm;
            const int bsw  = (bcol >> 2) & 3;
            FragU fb;
            fb.q[0] = *(const U4*)&Bs[bcol][(hf ^ bsw) << 2];
            fb.q[1] = *(const U4*)&Bs[bcol][((2 + hf) ^ bsw) << 2];
            acc[t] = __builtin_amdgcn_wmma_f32_16x16x32_bf16(
                false, fa.v, false, fb.v, (short)0, acc[t], false, false);
        }
        __syncthreads();
    };

    const int  kmain = K & ~31;          // full 32-wide K tiles
    const bool kvec4 = ((K & 3) == 0);   // A rows 16B-aligned -> float4 path

    for (int k0 = 0; k0 < kmain; k0 += 32) {
        if (kvec4) {
            // ---- A stage: 4x float4 loads, 2 packed u32 -> ds_store_b64
#pragma unroll
            for (int i = 0; i < 4; ++i) {
                int idx = i * 128 + tid;          // 512 k-quads
                int q = idx & 7, r = idx >> 3;
                const float* ap = A + (size_t)(m0 + r) * K + k0 + 4 * q;
                F4 v = *(const F4*)ap;
                unsigned int* dst =
                    &As[r][(((q >> 1) ^ ((r >> 2) & 3)) << 2) + ((2 * q) & 3)];
                dst[0] = packbf(v.x, v.y);
                dst[1] = packbf(v.z, v.w);
                __builtin_prefetch(ap + 32, 0, 1);   // next K tile
            }
        } else {
            // ---- A stage: unconditional paired b32 loads (odd K, full tile)
#pragma unroll
            for (int i = 0; i < 8; ++i) {
                int idx = i * 128 + tid;          // 1024 k-pairs
                int kp = idx & 15, r = idx >> 4;
                const float* ap = A + (size_t)(m0 + r) * K + k0 + 2 * kp;
                As[r][(((kp >> 2) ^ ((r >> 2) & 3)) << 2) + (kp & 3)] =
                    packbf(ap[0], ap[1]);
                __builtin_prefetch(ap + 32, 0, 1);
            }
        }
        // ---- B stage (column-major in LDS), branch-free
#pragma unroll
        for (int i = 0; i < 8; ++i) {
            int idx = i * 128 + tid;
            int cc = idx & 63, kp = idx >> 6;
            int gc = n0 + cc, k = k0 + 2 * kp;
            float v0, v1;
            if (TRANSW) {
                const float* wp = W + (size_t)gc * K + k;
                v0 = wp[0];
                v1 = wp[1];
                __builtin_prefetch(wp + 32, 0, 1);
            } else {
                v0 = W[(size_t)k * N + gc];
                v1 = W[(size_t)(k + 1) * N + gc];
                __builtin_prefetch(W + (size_t)(k + 32) * N + gc, 0, 1);
            }
            Bs[cc][(((kp >> 2) ^ ((cc >> 2) & 3)) << 2) + (kp & 3)] = packbf(v0, v1);
        }
        mma_step();
    }

    if (kmain < K) {  // uniform ragged-K tail (K in {147,195,259})
#pragma unroll
        for (int i = 0; i < 8; ++i) {
            int idx = i * 128 + tid;
            int kp = idx & 15, r = idx >> 4;
            int k = kmain + 2 * kp;
            const float* ap = A + (size_t)(m0 + r) * K + k;
            float v0 = (k < K) ? ap[0] : 0.0f;
            float v1 = (k + 1 < K) ? ap[1] : 0.0f;
            As[r][(((kp >> 2) ^ ((r >> 2) & 3)) << 2) + (kp & 3)] = packbf(v0, v1);
        }
#pragma unroll
        for (int i = 0; i < 8; ++i) {
            int idx = i * 128 + tid;
            int cc = idx & 63, kp = idx >> 6;
            int gc = n0 + cc, k = kmain + 2 * kp;
            float v0 = 0.0f, v1 = 0.0f;
            if (TRANSW) {
                const float* wp = W + (size_t)gc * K + k;
                if (k < K)     v0 = wp[0];
                if (k + 1 < K) v1 = wp[1];
            } else {
                if (k < K)     v0 = W[(size_t)k * N + gc];
                if (k + 1 < K) v1 = W[(size_t)(k + 1) * N + gc];
            }
            Bs[cc][(((kp >> 2) ^ ((cc >> 2) & 3)) << 2) + (kp & 3)] = packbf(v0, v1);
        }
        mma_step();
    }

    // ---- epilogue (M,N multiples of 64 -> unguarded stores)
#pragma unroll
    for (int t = 0; t < 4; ++t) {
        const int col = n0 + t * 16 + lm;
        const float bv = bias ? bias[col] : 0.0f;
#pragma unroll
        for (int r = 0; r < 8; ++r) {
            int row = m0 + wv * 16 + hf * 8 + r;
            float v = acc[t][r] + bv;
            if (ACCUM) v += C[(size_t)row * N + col];
            if (RELU) v = v > 0.0f ? v : 0.0f;
            C[(size_t)row * N + col] = v;
        }
    }
}

// ---------------------------------------------------------------------------
// Gather / concat / reduction kernels
// ---------------------------------------------------------------------------
__global__ void k_concat_edge1(const float* __restrict__ x, const int* __restrict__ row,
                               const int* __restrict__ col, const float* __restrict__ ea,
                               const float* __restrict__ u3, const int* __restrict__ batch,
                               float* __restrict__ out) {  // [NE,147]
    int e = blockIdx.x, c = threadIdx.x;
    if (c >= 147) return;
    float v;
    if (c < 64)        v = x[(size_t)row[e] * VIN + c];
    else if (c < 128)  v = x[(size_t)col[e] * VIN + (c - 64)];
    else if (c < 144)  v = ea[(size_t)e * EIN + (c - 128)];
    else               v = u3[(size_t)batch[row[e]] * UIN + (c - 144)];
    out[(size_t)e * 147 + c] = v;
}

__global__ void k_concat_edge2(const float* __restrict__ x, const int* __restrict__ row,
                               const int* __restrict__ col, const float* __restrict__ ea,
                               const float* __restrict__ u128, const int* __restrict__ batch,
                               float* __restrict__ out) {  // [NE,512]
    int e = blockIdx.x, c = threadIdx.x;
    float v;
    if (c < 128)       v = x[(size_t)row[e] * HID + c];
    else if (c < 256)  v = x[(size_t)col[e] * HID + (c - 128)];
    else if (c < 384)  v = ea[(size_t)e * HID + (c - 256)];
    else               v = u128[(size_t)batch[row[e]] * HID + (c - 384)];
    out[(size_t)e * 512 + c] = v;
}

__global__ void k_seg_sum_edge(const float* __restrict__ ea, const int* __restrict__ col,
                               float* __restrict__ agg) {
    int e = blockIdx.x, c = threadIdx.x;
    atomicAdd(&agg[(size_t)col[e] * HID + c], ea[(size_t)e * HID + c]);
}

__global__ void k_concat_node1(const float* __restrict__ x, const float* __restrict__ agg,
                               const float* __restrict__ u3, const int* __restrict__ batch,
                               float* __restrict__ out) {  // [NN,195]
    int n = blockIdx.x, c = threadIdx.x;
    if (c >= 195) return;
    float v;
    if (c < 64)        v = x[(size_t)n * VIN + c];
    else if (c < 192)  v = agg[(size_t)n * HID + (c - 64)];
    else               v = u3[(size_t)batch[n] * UIN + (c - 192)];
    out[(size_t)n * 195 + c] = v;
}

__global__ void k_concat_node2(const float* __restrict__ x, const float* __restrict__ agg,
                               const float* __restrict__ u128, const int* __restrict__ batch,
                               float* __restrict__ out) {  // [NN,384]
    int n = blockIdx.x, c = threadIdx.x;
    float v;
    if (c < 128)       v = x[(size_t)n * HID + c];
    else if (c < 256)  v = agg[(size_t)n * HID + (c - 128)];
    else               v = u128[(size_t)batch[n] * HID + (c - 256)];
    out[(size_t)n * 384 + c] = v;
}

__global__ void k_concat_glob1(const float* __restrict__ u3, const float* __restrict__ gm1,
                               const float* __restrict__ gm2, float* __restrict__ out) { // [BG,259]
    int g = blockIdx.x, c = threadIdx.x;
    if (c >= 259) return;
    float v;
    if (c < 3)         v = u3[(size_t)g * UIN + c];
    else if (c < 131)  v = gm1[(size_t)g * HID + (c - 3)];
    else               v = gm2[(size_t)g * HID + (c - 131)];
    out[(size_t)g * 259 + c] = v;
}

__global__ void k_concat_glob2(const float* __restrict__ u128, const float* __restrict__ gm1,
                               const float* __restrict__ gm2, float* __restrict__ out) { // [BG,384]
    int g = blockIdx.x, c = threadIdx.x;
    float v;
    if (c < 128)       v = u128[(size_t)g * HID + c];
    else if (c < 256)  v = gm1[(size_t)g * HID + (c - 128)];
    else               v = gm2[(size_t)g * HID + (c - 256)];
    out[(size_t)g * 384 + c] = v;
}

// per-graph mean over `cnt` contiguous rows of width C
__global__ void k_graph_mean(const float* __restrict__ src, float* __restrict__ dst,
                             int cnt, int C) {
    int g = blockIdx.x, c = threadIdx.x;
    const float* p = src + (size_t)g * cnt * C + c;
    float s = 0.0f;
    for (int i = 0; i < cnt; ++i) s += p[(size_t)i * C];
    dst[(size_t)g * C + c] = s / (float)cnt;
}

__global__ void k_gnorm_var(const float* __restrict__ x, const float* __restrict__ mu,
                            const float* __restrict__ ms, float* __restrict__ var) {
    int g = blockIdx.x, c = threadIdx.x;
    float m = ms[c] * mu[(size_t)g * HID + c];
    float s = 0.0f;
    for (int i = 0; i < NPG; ++i) {
        float d = x[((size_t)g * NPG + i) * HID + c] - m;
        s += d * d;
    }
    var[(size_t)g * HID + c] = s / (float)NPG;
}

__global__ void k_gnorm_apply(float* __restrict__ x, const float* __restrict__ mu,
                              const float* __restrict__ var, const float* __restrict__ ms,
                              const float* __restrict__ w, const float* __restrict__ b) {
    int n = blockIdx.x, c = threadIdx.x, g = n >> 6;  // NPG == 64
    float d = x[(size_t)n * HID + c] - ms[c] * mu[(size_t)g * HID + c];
    x[(size_t)n * HID + c] = w[c] * d * rsqrtf(var[(size_t)g * HID + c] + 1e-5f) + b[c];
}

// -------------------- MPNN stage --------------------
__global__ void k_node_feat(const float* __restrict__ xg1, const float* __restrict__ u1,
                            const float* __restrict__ xg2, const float* __restrict__ u2,
                            float* __restrict__ out) {  // [2*BG, 256]
    int n = blockIdx.x, c = threadIdx.x;
    const float* xg = (n < BG) ? xg1 : xg2;
    const float* uu = (n < BG) ? u1 : u2;
    int bidx = (n < BG) ? n : n - BG;
    float v = (c < HID) ? xg[(size_t)bidx * HID + c] : uu[(size_t)bidx * HID + (c - HID)];
    out[(size_t)n * H2 + c] = v;
}

__global__ void k_enet1(const float* __restrict__ svhb, const float* __restrict__ sohb,
                        const float* __restrict__ W1, const float* __restrict__ b1,
                        float* __restrict__ r) {  // [4*BG, EHD]
    int e = blockIdx.x, h = threadIdx.x;
    float ef = (e < 3 * BG) ? svhb[e & (BG - 1)] : sohb[e - 3 * BG];
    float v = ef * W1[h] + b1[h];
    r[(size_t)e * EHD + h] = v > 0.0f ? v : 0.0f;
}

__global__ void k_xsrc(const float* __restrict__ h, float* __restrict__ xsrc) {  // [4*BG,256]
    int e = blockIdx.x, c = threadIdx.x;
    int s = (e < 2 * BG) ? e : e - 2 * BG;  // src = [a, a+nb, c]
    xsrc[(size_t)e * H2 + c] = h[(size_t)s * H2 + c];
}

__global__ void k_build_z(const float* __restrict__ r, const float* __restrict__ xsrc,
                          float* __restrict__ z) {  // [4*BG, EHD*H2]
    int e = blockIdx.x, h = blockIdx.y, c = threadIdx.x;
    z[((size_t)e * EHD + h) * H2 + c] = r[(size_t)e * EHD + h] * xsrc[(size_t)e * H2 + c];
}

__global__ void k_seg_sum2(const float* __restrict__ msg, float* __restrict__ agg2) {
    int e = blockIdx.x, c = threadIdx.x;
    int d = (e < BG) ? e + BG : (e < 2 * BG) ? e - BG : e - 2 * BG;  // dst = [a+nb, a, c]
    atomicAdd(&agg2[(size_t)d * H2 + c], msg[(size_t)e * H2 + c]);
}

__global__ void k_add_relu(const float* __restrict__ a, const float* __restrict__ b,
                           float* __restrict__ o) {
    size_t i = (size_t)blockIdx.x * blockDim.x + threadIdx.x;
    float v = a[i] + b[i];
    o[i] = v > 0.0f ? v : 0.0f;
}

__global__ void k_gru(const float* __restrict__ gx, const float* __restrict__ gh,
                      const float* __restrict__ hid, float* __restrict__ xo) {
    int i = blockIdx.x, c = threadIdx.x;
    size_t b3 = (size_t)i * 3 * H2;
    float r = 1.0f / (1.0f + __expf(-(gx[b3 + c] + gh[b3 + c])));
    float z = 1.0f / (1.0f + __expf(-(gx[b3 + H2 + c] + gh[b3 + H2 + c])));
    float n = tanhf(gx[b3 + 2 * H2 + c] + r * gh[b3 + 2 * H2 + c]);
    float h = hid[(size_t)i * H2 + c];
    xo[(size_t)i * H2 + c] = (1.0f - z) * n + z * h;
}

__global__ void k_build_xg(const float* __restrict__ xf, float* __restrict__ xg) { // [BG,512]
    int b = blockIdx.x, c = threadIdx.x;
    float v = (c < H2) ? xf[(size_t)b * H2 + c] : xf[(size_t)(BG + b) * H2 + (c - H2)];
    xg[(size_t)b * 2 * H2 + c] = v;
}

__global__ void k_scale_T(const float* __restrict__ xg, const float* __restrict__ Tx,
                          float* __restrict__ o) {
    int b = blockIdx.x, c = threadIdx.x;
    o[(size_t)b * 2 * H2 + c] = xg[(size_t)b * 2 * H2 + c] * (Tx[b] + 273.15f);
}

__global__ void k_head_dot(const float* __restrict__ in, const float* __restrict__ W,
                           const float* __restrict__ b, float* __restrict__ out) {
    __shared__ float sm[128];
    int g = blockIdx.x, t = threadIdx.x;
    sm[t] = in[(size_t)g * HID + t] * W[t];
    __syncthreads();
    for (int s = 64; s > 0; s >>= 1) {
        if (t < s) sm[t] += sm[t + s];
        __syncthreads();
    }
    if (t == 0) out[g] = sm[0] + b[0];
}

__global__ void k_final(const float* __restrict__ ha, const float* __restrict__ hb,
                        const float* __restrict__ hc, const float* __restrict__ Tx,
                        float* __restrict__ out) {
    int i = blockIdx.x * blockDim.x + threadIdx.x;
    if (i < BG) {
        float T = Tx[i] + 273.15f;
        out[i] = ha[i] + hb[i] / T + hc[i];
    }
}

// ---------------------------------------------------------------------------
// Host orchestration
// ---------------------------------------------------------------------------
static void launch_gemm(const float* A, const float* W, const float* bias, float* C,
                        int M, int N, int K, int relu, int transw, int accum,
                        hipStream_t s) {
    dim3 g((M + 63) / 64, (N + 63) / 64), b(128);
    if (transw)
        hipLaunchKernelGGL((gemm_bf16_wmma<1, 0, 0>), g, b, 0, s, A, W, bias, C, M, N, K);
    else if (accum)
        hipLaunchKernelGGL((gemm_bf16_wmma<0, 0, 1>), g, b, 0, s, A, W, bias, C, M, N, K);
    else if (relu)
        hipLaunchKernelGGL((gemm_bf16_wmma<0, 1, 0>), g, b, 0, s, A, W, bias, C, M, N, K);
    else
        hipLaunchKernelGGL((gemm_bf16_wmma<0, 0, 0>), g, b, 0, s, A, W, bias, C, M, N, K);
}

extern "C" void kernel_launch(void* const* d_in, const int* in_sizes, int n_in,
                              void* d_out, int out_size, void* d_ws, size_t ws_size,
                              hipStream_t stream) {
    (void)in_sizes; (void)n_in; (void)out_size; (void)ws_size;

    // ---- inputs (setup_inputs dict order, params flattened in insertion order)
    const float* sv_x   = (const float*)d_in[0];
    const int*   sv_row = (const int*)d_in[1];
    const int*   sv_col = sv_row + NE;
    const float* sv_ea  = (const float*)d_in[2];
    const int*   sv_b   = (const int*)d_in[3];
    const float* sv_u   = (const float*)d_in[4];
    const float* sv_hb  = (const float*)d_in[5];
    const float* so_x   = (const float*)d_in[6];
    const int*   so_row = (const int*)d_in[7];
    const int*   so_col = so_row + NE;
    const float* so_ea  = (const float*)d_in[8];
    const int*   so_b   = (const int*)d_in[9];
    const float* so_u   = (const float*)d_in[10];
    const float* so_hb  = (const float*)d_in[11];
    const float* Tx     = (const float*)d_in[12];
    const float* P[73];
    for (int i = 13; i < 73; ++i) P[i] = (const float*)d_in[i];

    // ---- workspace carve-out
    size_t off = 0;
    auto alloc = [&](size_t nf) {
        float* p = (float*)((char*)d_ws + off);
        off += ((nf * sizeof(float) + 255) / 256) * 256;
        return p;
    };
    float* big0   = alloc((size_t)NE * 512);      // edge_in / node_in / glob_in / z
    float* ebuf   = alloc((size_t)NE * HID);      // MLP hidden (edge- or node-sized)
    float* ea_cur = alloc((size_t)NE * HID);
    float* agg    = alloc((size_t)NN * HID);
    float* xb     = alloc((size_t)NN * HID);
    float* gm1    = alloc((size_t)BG * HID);
    float* gm2    = alloc((size_t)BG * HID);
    float* mu     = alloc((size_t)BG * HID);
    float* var    = alloc((size_t)BG * HID);
    float* uh     = alloc((size_t)BG * HID);
    float* xg1    = alloc((size_t)BG * HID);
    float* u1     = alloc((size_t)BG * HID);
    float* xg2    = alloc((size_t)BG * HID);
    float* u2     = alloc((size_t)BG * HID);
    float* nfeat  = alloc((size_t)2 * BG * H2);
    float* hbuf   = alloc((size_t)2 * BG * H2);
    float* renet  = alloc((size_t)4 * BG * EHD);
    float* xsrc   = alloc((size_t)4 * BG * H2);
    float* msg    = alloc((size_t)4 * BG * H2);
    float* agg2   = alloc((size_t)2 * BG * H2);
    float* xr     = alloc((size_t)2 * BG * H2);
    float* x2     = alloc((size_t)2 * BG * H2);
    float* gx     = alloc((size_t)2 * BG * 3 * H2);
    float* gh     = alloc((size_t)2 * BG * 3 * H2);
    float* xfin   = alloc((size_t)2 * BG * H2);
    float* xg     = alloc((size_t)BG * 2 * H2);
    float* xgT    = alloc((size_t)BG * 2 * H2);
    float* t1     = alloc((size_t)BG * HID);
    float* t2     = alloc((size_t)BG * HID);
    float* ha     = alloc(BG);
    float* hb     = alloc(BG);
    float* hc     = alloc(BG);

    // ---- one Meta-layer branch (shared params, two graph inputs)
    auto branch = [&](const float* X, const int* ROW, const int* COL, const float* EA,
                      const int* BATCH, const float* U3, float* xg_out, float* u_out) {
        // g1 edge model
        k_concat_edge1<<<NE, 160, 0, stream>>>(X, ROW, COL, EA, U3, BATCH, big0);
        launch_gemm(big0, P[13], P[14], ebuf, NE, HID, 147, 1, 0, 0, stream);
        launch_gemm(ebuf, P[15], P[16], ea_cur, NE, HID, HID, 0, 0, 0, stream);
        // g1 node model
        hipMemsetAsync(agg, 0, (size_t)NN * HID * sizeof(float), stream);
        k_seg_sum_edge<<<NE, HID, 0, stream>>>(ea_cur, COL, agg);
        k_concat_node1<<<NN, 224, 0, stream>>>(X, agg, U3, BATCH, big0);
        launch_gemm(big0, P[17], P[18], ebuf, NN, HID, 195, 1, 0, 0, stream);
        launch_gemm(ebuf, P[19], P[20], xb, NN, HID, HID, 0, 0, 0, stream);
        // g1 global model (per-graph counts are exact: 64 nodes, 128 contiguous edges)
        k_graph_mean<<<BG, HID, 0, stream>>>(xb, gm1, NPG, HID);
        k_graph_mean<<<BG, HID, 0, stream>>>(ea_cur, gm2, EPG, HID);
        k_concat_glob1<<<BG, 288, 0, stream>>>(U3, gm1, gm2, big0);
        launch_gemm(big0, P[21], P[22], uh, BG, HID, 259, 1, 0, 0, stream);
        launch_gemm(uh, P[23], P[24], u_out, BG, HID, HID, 0, 0, 0, stream);
        // GraphNorm 1
        k_graph_mean<<<BG, HID, 0, stream>>>(xb, mu, NPG, HID);
        k_gnorm_var<<<BG, HID, 0, stream>>>(xb, mu, P[39], var);
        k_gnorm_apply<<<NN, HID, 0, stream>>>(xb, mu, var, P[39], P[37], P[38]);
        // g2 edge model
        k_concat_edge2<<<NE, 512, 0, stream>>>(xb, ROW, COL, ea_cur, u_out, BATCH, big0);
        launch_gemm(big0, P[25], P[26], ebuf, NE, HID, 512, 1, 0, 0, stream);
        launch_gemm(ebuf, P[27], P[28], ea_cur, NE, HID, HID, 0, 0, 0, stream);
        // g2 node model
        hipMemsetAsync(agg, 0, (size_t)NN * HID * sizeof(float), stream);
        k_seg_sum_edge<<<NE, HID, 0, stream>>>(ea_cur, COL, agg);
        k_concat_node2<<<NN, 384, 0, stream>>>(xb, agg, u_out, BATCH, big0);
        launch_gemm(big0, P[29], P[30], ebuf, NN, HID, 384, 1, 0, 0, stream);
        launch_gemm(ebuf, P[31], P[32], xb, NN, HID, HID, 0, 0, 0, stream);
        // g2 global model
        k_graph_mean<<<BG, HID, 0, stream>>>(xb, gm1, NPG, HID);
        k_graph_mean<<<BG, HID, 0, stream>>>(ea_cur, gm2, EPG, HID);
        k_concat_glob2<<<BG, 384, 0, stream>>>(u_out, gm1, gm2, big0);
        launch_gemm(big0, P[33], P[34], uh, BG, HID, 384, 1, 0, 0, stream);
        launch_gemm(uh, P[35], P[36], u_out, BG, HID, HID, 0, 0, 0, stream);
        // GraphNorm 2 + graph readout
        k_graph_mean<<<BG, HID, 0, stream>>>(xb, mu, NPG, HID);
        k_gnorm_var<<<BG, HID, 0, stream>>>(xb, mu, P[42], var);
        k_gnorm_apply<<<NN, HID, 0, stream>>>(xb, mu, var, P[42], P[40], P[41]);
        k_graph_mean<<<BG, HID, 0, stream>>>(xb, xg_out, NPG, HID);
    };

    branch(sv_x, sv_row, sv_col, sv_ea, sv_b, sv_u, xg1, u1);
    branch(so_x, so_row, so_col, so_ea, so_b, so_u, xg2, u2);

    // ---- MPNN conv over the 2B-node pair graph
    k_node_feat<<<2 * BG, H2, 0, stream>>>(xg1, u1, xg2, u2, nfeat);
    launch_gemm(nfeat, P[43], P[44], hbuf, 2 * BG, H2, H2, 1, 0, 0, stream);  // h = relu(proj)
    k_enet1<<<4 * BG, EHD, 0, stream>>>(sv_hb, so_hb, P[45], P[46], renet);
    k_xsrc<<<4 * BG, H2, 0, stream>>>(hbuf, xsrc);
    // msg = (r ⊗ x_src) @ W2_flat[8192,256] + x_src @ b2_flat[256,256]
    k_build_z<<<dim3(4 * BG, EHD), H2, 0, stream>>>(renet, xsrc, big0);
    launch_gemm(big0, P[47], nullptr, msg, 4 * BG, H2, EHD * H2, 0, 0, 0, stream);
    launch_gemm(xsrc, P[48], nullptr, msg, 4 * BG, H2, H2, 0, 0, 1, stream);
    hipMemsetAsync(agg2, 0, (size_t)2 * BG * H2 * sizeof(float), stream);
    k_seg_sum2<<<4 * BG, H2, 0, stream>>>(msg, agg2);
    launch_gemm(hbuf, P[49], P[50], xr, 2 * BG, H2, H2, 0, 0, 0, stream);     // x@root + conv_b
    k_add_relu<<<2 * BG, H2, 0, stream>>>(agg2, xr, x2);
    // GRU step
    launch_gemm(x2,   P[51], P[53], gx, 2 * BG, 3 * H2, H2, 0, 1, 0, stream); // x @ W_ih^T + b_ih
    launch_gemm(hbuf, P[52], P[54], gh, 2 * BG, 3 * H2, H2, 0, 1, 0, stream); // h @ W_hh^T + b_hh
    k_gru<<<2 * BG, H2, 0, stream>>>(gx, gh, hbuf, xfin);
    k_build_xg<<<BG, 2 * H2, 0, stream>>>(xfin, xg);
    k_scale_T<<<BG, 2 * H2, 0, stream>>>(xg, Tx, xgT);

    // ---- heads
    auto head = [&](int base, const float* in, float* outv) {
        launch_gemm(in, P[base], P[base + 1], t1, BG, HID, 4 * HID, 1, 0, 0, stream);
        launch_gemm(t1, P[base + 2], P[base + 3], t2, BG, HID, HID, 1, 0, 0, stream);
        k_head_dot<<<BG, HID, 0, stream>>>(t2, P[base + 4], P[base + 5], outv);
    };
    head(55, xg, ha);
    head(61, xg, hb);
    head(67, xgT, hc);
    k_final<<<(BG + 255) / 256, 256, 0, stream>>>(ha, hb, hc, Tx, (float*)d_out);
}
